// MPNNGNN_11278584119662
// MI455X (gfx1250) — matrix operand order repaired and checked by reading, not compiled, not run on hardware
//
#include <hip/hip_runtime.h>
#include <hip/hip_bf16.h>
#include <stdint.h>

#define N_IN   32   // NODE_IN
#define E_IN   16   // EDGE_IN
#define HF     32   // H
#define EHF    128  // EH
#define N_OUT  64   // NODE_OUT
#define STEPS  6

typedef __attribute__((ext_vector_type(16))) __bf16 v16bf;
typedef __attribute__((ext_vector_type(8)))  __bf16 v8bf;
typedef __attribute__((ext_vector_type(8)))  float  v8f;

// ---------------------------------------------------------------- utilities
__global__ void k_zero(uint32_t* __restrict__ p, int n) {
    int i = blockIdx.x * blockDim.x + threadIdx.x;
    if (i < n) p[i] = 0u;
}

__global__ void k_copy_f32(const float* __restrict__ s, float* __restrict__ d, int n) {
    int i = blockIdx.x * blockDim.x + threadIdx.x;
    if (i < n) d[i] = s[i];
}

// edge_w2 (EHF x 1024) fp32 -> transposed bf16 (1024 x EHF) for WMMA B-fragments
__global__ void k_w2t(const float* __restrict__ w2, __bf16* __restrict__ w2t, int n) {
    int i = blockIdx.x * blockDim.x + threadIdx.x;       // linear over w2 (coalesced read)
    if (i >= n) return;
    int o = i & (HF * HF - 1);                           // col in 0..1023
    int k = i >> 10;                                     // row in 0..127
    w2t[(size_t)o * EHF + k] = (__bf16)w2[i];
}

// in-degree count + reciprocal
__global__ void k_count(const int* __restrict__ dst, int* __restrict__ cnt, int E) {
    int i = blockIdx.x * blockDim.x + threadIdx.x;
    if (i < E)
        __hip_atomic_fetch_add(&cnt[dst[i]], 1, __ATOMIC_RELAXED, __HIP_MEMORY_SCOPE_AGENT);
}
__global__ void k_inv(const int* __restrict__ cnt, float* __restrict__ inv, int N) {
    int i = blockIdx.x * blockDim.x + threadIdx.x;
    if (i < N) inv[i] = 1.0f / fmaxf((float)cnt[i], 1.0f);
}

// ------------------------------------------------- node projection MLP (32->32->32)
__global__ __launch_bounds__(256)
void k_proj(const float* __restrict__ nf,
            const float* __restrict__ w1, const float* __restrict__ b1,
            const float* __restrict__ w2, const float* __restrict__ b2,
            float* __restrict__ hid, int N) {
    __shared__ float sW1[HF * HF], sW2[HF * HF], sB1[HF], sB2[HF];
    int t = threadIdx.x;
    for (int i = t; i < HF * HF; i += 256) { sW1[i] = w1[i]; sW2[i] = w2[i]; }
    if (t < HF) { sB1[t] = b1[t]; sB2[t] = b2[t]; }
    __syncthreads();
    int node = blockIdx.x * 8 + (t >> 5);
    if (node >= N) return;
    int l = t & 31;
    float x = nf[(size_t)node * N_IN + l];
    float a = sB1[l];
    #pragma unroll 8
    for (int k = 0; k < N_IN; ++k) a += __shfl(x, k, 32) * sW1[k * HF + l];
    a = fmaxf(a, 0.0f);
    float h = sB2[l];
    #pragma unroll 8
    for (int k = 0; k < HF; ++k) h += __shfl(a, k, 32) * sW2[k * HF + l];
    hid[(size_t)node * HF + l] = h;
}

// ------------------------------------------------- fused edge network (WMMA bf16)
// ew = relu(ef@W1+b1) computed into LDS (bf16), then (64x128)@(128x1024) via
// v_wmma_f32_16x16x32_bf16 with bias-seeded accumulators, stored bf16.
#define ETILE 64
__global__ __launch_bounds__(256)
void k_edge(const float* __restrict__ ef,
            const float* __restrict__ w1, const float* __restrict__ b1,
            const __bf16* __restrict__ w2t, const float* __restrict__ b2,
            __bf16* __restrict__ Wout, int E) {
    __shared__ __bf16 sA[ETILE * EHF];   // 16 KB, row-major [edge][k]
    int t = threadIdx.x;
    int e0 = blockIdx.x * ETILE;
    const bool full = (e0 + ETILE) <= E;   // uniform: all rows valid

    // phase 1: edge hidden layer (VALU, tiny: 16 FMAs/value)
    for (int i = t; i < ETILE * EHF; i += 256) {
        int e = i >> 7, o = i & 127;
        float s = 0.0f;
        if (e0 + e < E) {
            s = b1[o];
            const float* efr = ef + (size_t)(e0 + e) * E_IN;
            #pragma unroll
            for (int k = 0; k < E_IN; ++k) s += efr[k] * w1[k * EHF + o];
            s = fmaxf(s, 0.0f);
        }
        sA[i] = (__bf16)s;
    }
    __syncthreads();

    // phase 2: WMMA GEMM. 8 waves; each wave owns n-tiles w, w+8, ...
    int w = t >> 5, l = t & 31;
    int lmod = l & 15, lhi = l >> 4;
    for (int nt = w; nt < (HF * HF) / 16; nt += 8) {
        int col = nt * 16 + lmod;
        // B 32x16 bf16 fragment: lane = col (split at lane16 for K halves),
        // 16 contiguous K values per lane from transposed weights.
        const __bf16* brow = w2t + (size_t)col * EHF + lhi * 16;
        v16bf B0 = *(const v16bf*)(brow);
        v16bf B1 = *(const v16bf*)(brow + 32);
        v16bf B2 = *(const v16bf*)(brow + 64);
        v16bf B3 = *(const v16bf*)(brow + 96);
        float bias = b2[col];
        for (int ms = 0; ms < 4; ++ms) {
            // seed accumulator with the (column-constant) bias: no post-GEMM add
            v8f acc;
            #pragma unroll
            for (int i = 0; i < 8; ++i) acc[i] = bias;
            const __bf16* arow = sA + (ms * 16 + lmod) * EHF + lhi * 16;
            #pragma unroll
            for (int kt = 0; kt < 4; ++kt) {
                // A 16x32 bf16 layout: elems 0..15 = K[kb..kb+15] (kb includes lane-half*16)
                v8bf lo = *(const v8bf*)(arow + kt * 32);
                v8bf hi = *(const v8bf*)(arow + kt * 32 + 8);
                v16bf A;
                #pragma unroll
                for (int i = 0; i < 8; ++i) { A[i] = lo[i]; A[i + 8] = hi[i]; }
                v16bf Bk = (kt == 0) ? B0 : (kt == 1) ? B1 : (kt == 2) ? B2 : B3;
                acc = __builtin_amdgcn_wmma_f32_16x16x32_bf16(
                        false, A, false, Bk, (short)0, acc, false, false);
            }
            int rbase = e0 + ms * 16 + lhi * 8;
            __bf16* outp = Wout + (size_t)rbase * (HF * HF) + col;
            if (full) {
                #pragma unroll
                for (int r = 0; r < 8; ++r)
                    outp[(size_t)r * (HF * HF)] = (__bf16)acc[r];
            } else {
                #pragma unroll
                for (int r = 0; r < 8; ++r)
                    if (rbase + r < E)
                        outp[(size_t)r * (HF * HF)] = (__bf16)acc[r];
            }
        }
    }
}

// ------------------------------------------------- per-edge matvec + scatter-add
// Lane layout: p=l&15 owns output pair (2p, 2p+1); halves q=l>>4 cover even/odd
// h rows -> every wave load request is a fully-coalesced 128B (32 x b32).
__global__ __launch_bounds__(256)
void k_msg(const float* __restrict__ hid, const __bf16* __restrict__ W,
           const int* __restrict__ src, const int* __restrict__ dst,
           float* __restrict__ agg, int E) {
    int t = threadIdx.x;
    int e = blockIdx.x * 8 + (t >> 5);
    if (e >= E) return;
    int l = t & 31;
    int p = l & 15, q = l >> 4;
    int s = src[e], d = dst[e];
    float x = hid[(size_t)s * HF + l];
    const uint32_t* Wr = (const uint32_t*)(W + (size_t)e * (HF * HF)) + p;
    float a0 = 0.0f, a1 = 0.0f;
    #pragma unroll 4
    for (int h2 = 0; h2 < 16; ++h2) {
        int h = 2 * h2 + q;
        uint32_t u = Wr[(size_t)h * 16];                       // W[e][h][2p..2p+1]
        float w0 = __builtin_bit_cast(float, u << 16);          // bf16 -> f32 exact
        float w1 = __builtin_bit_cast(float, u & 0xffff0000u);
        float xh = __shfl(x, h, 32);
        a0 += xh * w0;
        a1 += xh * w1;
    }
    // combine even-h half with odd-h half
    a0 += __shfl_xor(a0, 16, 32);
    a1 += __shfl_xor(a1, 16, 32);
    float v = q ? a1 : a0;                                      // lane -> output 2p+q
    __hip_atomic_fetch_add(&agg[(size_t)d * HF + 2 * p + q], v,
                           __ATOMIC_RELAXED, __HIP_MEMORY_SCOPE_AGENT);
}

// ------------------------------------------------- mean + ReLU + GRU (in-place hid)
__global__ __launch_bounds__(256)
void k_gru(const float* __restrict__ agg, const float* __restrict__ inv,
           const float* __restrict__ conv_bias,
           const float* __restrict__ w_ih, const float* __restrict__ b_ih,
           const float* __restrict__ w_hh, const float* __restrict__ b_hh,
           float* __restrict__ hid, int N) {
    __shared__ float sWi[HF * 3 * HF], sWh[HF * 3 * HF];
    __shared__ float sBi[3 * HF], sBh[3 * HF], sCb[HF];
    int t = threadIdx.x;
    for (int i = t; i < HF * 3 * HF; i += 256) { sWi[i] = w_ih[i]; sWh[i] = w_hh[i]; }
    for (int i = t; i < 3 * HF; i += 256) { sBi[i] = b_ih[i]; sBh[i] = b_hh[i]; }
    if (t < HF) sCb[t] = conv_bias[t];
    __syncthreads();
    int node = blockIdx.x * 8 + (t >> 5);
    if (node >= N) return;
    int l = t & 31;
    float x  = fmaxf(agg[(size_t)node * HF + l] * inv[node] + sCb[l], 0.0f);
    float hp = hid[(size_t)node * HF + l];
    float gir = sBi[l], giz = sBi[HF + l], gin = sBi[2 * HF + l];
    float ghr = sBh[l], ghz = sBh[HF + l], ghn = sBh[2 * HF + l];
    #pragma unroll 4
    for (int k = 0; k < HF; ++k) {
        float xk = __shfl(x, k, 32), hk = __shfl(hp, k, 32);
        const float* wi = sWi + k * 3 * HF;
        const float* wh = sWh + k * 3 * HF;
        gir += xk * wi[l];          ghr += hk * wh[l];
        giz += xk * wi[HF + l];     ghz += hk * wh[HF + l];
        gin += xk * wi[2 * HF + l]; ghn += hk * wh[2 * HF + l];
    }
    float r = 1.0f / (1.0f + __expf(-(gir + ghr)));
    float z = 1.0f / (1.0f + __expf(-(giz + ghz)));
    float n = tanhf(gin + r * ghn);
    hid[(size_t)node * HF + l] = (1.0f - z) * n + z * hp;
}

// ------------------------------------------------- decoder MLP (32->32->64)
__global__ __launch_bounds__(256)
void k_dec(const float* __restrict__ hid,
           const float* __restrict__ w1, const float* __restrict__ b1,
           const float* __restrict__ w2, const float* __restrict__ b2,
           float* __restrict__ out, int N) {
    __shared__ float sW1[HF * HF], sW2[HF * N_OUT], sB1[HF], sB2[N_OUT];
    int t = threadIdx.x;
    for (int i = t; i < HF * HF; i += 256) sW1[i] = w1[i];
    for (int i = t; i < HF * N_OUT; i += 256) sW2[i] = w2[i];
    if (t < HF) sB1[t] = b1[t];
    if (t < N_OUT) sB2[t] = b2[t];
    __syncthreads();
    int node = blockIdx.x * 8 + (t >> 5);
    if (node >= N) return;
    int l = t & 31;
    float h = hid[(size_t)node * HF + l];
    float a = sB1[l];
    #pragma unroll 8
    for (int k = 0; k < HF; ++k) a += __shfl(h, k, 32) * sW1[k * HF + l];
    a = fmaxf(a, 0.0f);
    float o0 = sB2[l], o1 = sB2[HF + l];
    #pragma unroll 8
    for (int k = 0; k < HF; ++k) {
        float ak = __shfl(a, k, 32);
        o0 += ak * sW2[k * N_OUT + l];
        o1 += ak * sW2[k * N_OUT + HF + l];
    }
    out[(size_t)node * N_OUT + l]      = o0;
    out[(size_t)node * N_OUT + HF + l] = o1;
}

// ---------------------------------------------------------------- launcher
extern "C" void kernel_launch(void* const* d_in, const int* in_sizes, int n_in,
                              void* d_out, int out_size, void* d_ws, size_t ws_size,
                              hipStream_t stream) {
    const float* nf        = (const float*)d_in[0];
    const float* ef        = (const float*)d_in[1];
    const int*   src       = (const int*)d_in[2];
    const int*   dst       = (const int*)d_in[3];
    const float* proj_w1   = (const float*)d_in[4];
    const float* proj_b1   = (const float*)d_in[5];
    const float* proj_w2   = (const float*)d_in[6];
    const float* proj_b2   = (const float*)d_in[7];
    const float* edge_w1   = (const float*)d_in[8];
    const float* edge_b1   = (const float*)d_in[9];
    const float* edge_w2   = (const float*)d_in[10];
    const float* edge_b2   = (const float*)d_in[11];
    const float* conv_bias = (const float*)d_in[12];
    const float* gru_w_ih  = (const float*)d_in[13];
    const float* gru_b_ih  = (const float*)d_in[14];
    const float* gru_w_hh  = (const float*)d_in[15];
    const float* gru_b_hh  = (const float*)d_in[16];
    const float* dec_w1    = (const float*)d_in[17];
    const float* dec_b1    = (const float*)d_in[18];
    const float* dec_w2    = (const float*)d_in[19];
    const float* dec_b2    = (const float*)d_in[20];

    const int E = in_sizes[2];
    const int N = in_sizes[0] / N_IN;

    // workspace carve-out (256B aligned)
    char* ws = (char*)d_ws;
    auto carve = [&](size_t bytes) -> char* {
        char* p = ws; ws += (bytes + 255) & ~(size_t)255; return p;
    };
    float*  hid  = (float*)carve((size_t)N * HF * sizeof(float));
    float*  agg  = (float*)carve((size_t)N * HF * sizeof(float));
    int*    cnt  = (int*)  carve((size_t)N * sizeof(int));
    float*  inv  = (float*)carve((size_t)N * sizeof(float));
    __bf16* w2t  = (__bf16*)carve((size_t)EHF * HF * HF * sizeof(__bf16));
    __bf16* Wbig = (__bf16*)carve((size_t)E * HF * HF * sizeof(__bf16));

    // one-time preprocessing
    k_w2t  <<<(EHF * HF * HF + 255) / 256, 256, 0, stream>>>(edge_w2, w2t, EHF * HF * HF);
    k_zero <<<(N + 255) / 256, 256, 0, stream>>>((uint32_t*)cnt, N);
    k_count<<<(E + 255) / 256, 256, 0, stream>>>(dst, cnt, E);
    k_inv  <<<(N + 255) / 256, 256, 0, stream>>>(cnt, inv, N);
    k_proj <<<(N + 7) / 8, 256, 0, stream>>>(nf, proj_w1, proj_b1, proj_w2, proj_b2, hid, N);
    k_edge <<<(E + ETILE - 1) / ETILE, 256, 0, stream>>>(ef, edge_w1, edge_b1, w2t, edge_b2, Wbig, E);

    // message-passing steps (node == hidden at msg time, so one buffer suffices)
    for (int s = 0; s < STEPS; ++s) {
        k_zero<<<((size_t)N * HF + 255) / 256, 256, 0, stream>>>((uint32_t*)agg, N * HF);
        k_msg <<<(E + 7) / 8, 256, 0, stream>>>(hid, Wbig, src, dst, agg, E);
        k_gru <<<(N + 7) / 8, 256, 0, stream>>>(agg, inv, conv_bias,
                                                gru_w_ih, gru_b_ih, gru_w_hh, gru_b_hh, hid, N);
    }

    if (out_size == N * N_OUT) {
        k_dec<<<(N + 7) / 8, 256, 0, stream>>>(hid, dec_w1, dec_b1, dec_w2, dec_b2,
                                               (float*)d_out, N);
    } else {
        k_copy_f32<<<((size_t)N * HF + 255) / 256, 256, 0, stream>>>(hid, (float*)d_out, N * HF);
    }
}
